// MolecularGNN_61521111547950
// MI455X (gfx1250) — matrix operand order repaired and checked by reading, not compiled, not run on hardware
//
#include <hip/hip_runtime.h>

#define DN 128
#define HN 8
#define NN 100000
#define EN 400000
#define GN 4096
#define ENF (EN + NN)

#define CDIV(a, b) (((a) + (b) - 1) / (b))

typedef __attribute__((ext_vector_type(16))) __bf16 v16bf;
typedef __attribute__((ext_vector_type(8)))  float  v8f;

static __device__ __forceinline__ unsigned short f2bf(float x) {
  unsigned u = __float_as_uint(x);
  u += 0x7fffu + ((u >> 16) & 1u);   // round-to-nearest-even
  return (unsigned short)(u >> 16);
}
static __device__ __forceinline__ float bf2f(unsigned short h) {
  return __uint_as_float(((unsigned)h) << 16);
}
// monotonic float <-> uint encoding for atomicMax on floats
static __device__ __forceinline__ unsigned fenc(float f) {
  unsigned u = __float_as_uint(f);
  return (u & 0x80000000u) ? ~u : (u | 0x80000000u);
}
static __device__ __forceinline__ float fdec(unsigned u) {
  unsigned v = (u & 0x80000000u) ? (u & 0x7fffffffu) : ~u;
  return __uint_as_float(v);
}

union BF16x16 { v16bf v; unsigned short s[16]; };

static __device__ __forceinline__ v8f wmma_bf16(v16bf a, v16bf b, v8f c) {
  // D = A(16x32 bf16) * B(32x16 bf16) + C(16x16 f32)
  return __builtin_amdgcn_wmma_f32_16x16x32_bf16(
      /*neg_a=*/false, a, /*neg_b=*/false, b,
      /*c_mod=*/(short)0, c, /*reuse_a=*/false, /*reuse_b=*/false);
}

// ---------------------------------------------------------------------------
// WMMA GEMM: out[M,128] = A[M,128] @ W[128,128], fp32 via bf16 hi/lo split.
// Optionally fuses per-head attention dots:
//   dotA[m,h] = sum_c out[m, h*16+c] * attA[h*16+c]   (and same for attB/dotB)
// Weights are staged in LDS pre-swizzled into B-fragment order (bf16 hi + lo).
// ---------------------------------------------------------------------------
__global__ __launch_bounds__(256) void gemm128_wmma_kernel(
    const float* __restrict__ A, long long M,
    const float* __restrict__ W,
    float* __restrict__ outFull,
    const float* __restrict__ attA, const float* __restrict__ attB,
    float* __restrict__ dotA, float* __restrict__ dotB)
{
  __shared__ unsigned short ldsHi[DN * DN];
  __shared__ unsigned short ldsLo[DN * DN];

  // Fill LDS with W in B-fragment order:
  // tile = ct*4+kt; within tile: lane*16 + j.
  // B (32x16, 16-bit): lane half selects K range: K = kt*32 + 16*(lane>>4) + j,
  // col = ct*16 + (lane&15).
  for (int i = threadIdx.x; i < DN * DN; i += 256) {
    int tile = i >> 9;
    int ln   = (i >> 4) & 31;
    int j    = i & 15;
    int ct = tile >> 2, kt = tile & 3;
    int col = ct * 16 + (ln & 15);
    int K   = kt * 32 + 16 * (ln >> 4) + j;
    float w = W[K * DN + col];
    unsigned short hh = f2bf(w);
    ldsHi[i] = hh;
    ldsLo[i] = f2bf(w - bf2f(hh));
  }
  __syncthreads();

  int wave = threadIdx.x >> 5;
  int lane = threadIdx.x & 31;
  int half = lane >> 4;
  int lm   = lane & 15;
  long long rowBase = (long long)blockIdx.x * 128 + wave * 16;

  // A fragments (16x32 bf16): lanes 0-15 and 16-31 both hold rows M=0..15;
  // element j maps to K = kt*32 + 16*half + j.
  v16bf aHi[4], aLo[4];
  {
    long long row = rowBase + lm;
    bool ok = (row < M);
    const float* Ar = A + (ok ? row : 0) * DN;
    for (int kt = 0; kt < 4; ++kt) {
      BF16x16 hi, lo;
      for (int jp = 0; jp < 8; ++jp) {
        int j = jp << 1;
        int K = kt * 32 + 16 * half + j;
        float v0 = 0.f, v1 = 0.f;
        if (ok) { float2 t = *(const float2*)(Ar + K); v0 = t.x; v1 = t.y; }
        unsigned short h0 = f2bf(v0), h1 = f2bf(v1);
        hi.s[j] = h0;              hi.s[j + 1] = h1;
        lo.s[j] = f2bf(v0 - bf2f(h0)); lo.s[j + 1] = f2bf(v1 - bf2f(h1));
      }
      aHi[kt] = hi.v;
      aLo[kt] = lo.v;
    }
  }

  for (int ct = 0; ct < 8; ++ct) {           // col tile == attention head
    v8f acc = {0.f, 0.f, 0.f, 0.f, 0.f, 0.f, 0.f, 0.f};
    for (int kt = 0; kt < 4; ++kt) {
      int off = ((ct * 4 + kt) << 9) + (lane << 4);
      v16bf bHi = *(const v16bf*)&ldsHi[off];
      v16bf bLo = *(const v16bf*)&ldsLo[off];
      acc = wmma_bf16(aHi[kt], bHi, acc);    // hi*hi
      acc = wmma_bf16(aLo[kt], bHi, acc);    // lo*hi
      acc = wmma_bf16(aHi[kt], bLo, acc);    // hi*lo
    }
    // D layout: VGPR r, lanes 0-15 -> row r, lanes 16-31 -> row r+8; col = lane&15.
    if (outFull) {
      for (int r = 0; r < 8; ++r) {
        long long row = rowBase + r + 8 * half;
        if (row < M) outFull[row * DN + ct * 16 + lm] = acc[r];
      }
    }
    if (dotA) {
      for (int r = 0; r < 8; ++r) {
        long long row = rowBase + r + 8 * half;
        float tA = acc[r] * attA[ct * 16 + lm];
        for (int o = 8; o >= 1; o >>= 1) tA += __shfl_xor(tA, o, 16);
        if (attB) {
          float tB = acc[r] * attB[ct * 16 + lm];
          for (int o = 8; o >= 1; o >>= 1) tB += __shfl_xor(tB, o, 16);
          if (lm == 0 && row < M) dotB[row * HN + ct] = tB;
        }
        if (lm == 0 && row < M) dotA[row * HN + ct] = tA;
      }
    }
  }
}

// ---------------------------------------------------------------------------
// Supporting streaming / scatter kernels
// ---------------------------------------------------------------------------
__global__ void zero_f32(float* p, long long n) {
  long long i = (long long)blockIdx.x * blockDim.x + threadIdx.x;
  if (i < n) p[i] = 0.f;
}

__global__ void embed_node_kernel(const float* __restrict__ x,
                                  const float* __restrict__ Wn,
                                  const float* __restrict__ bn,
                                  float* __restrict__ h, float* __restrict__ res) {
  long long i = (long long)blockIdx.x * blockDim.x + threadIdx.x;
  if (i >= (long long)NN * DN) return;
  int n = (int)(i >> 7), c = (int)(i & 127);
  float a = bn[c];
  const float* xr = x + (long long)n * 9;
  for (int k = 0; k < 9; ++k) a += xr[k] * Wn[k * DN + c];
  h[i] = a;
  res[i] = a;
}

__global__ void embed_edge_kernel(const float* __restrict__ ea,
                                  const float* __restrict__ We,
                                  const float* __restrict__ be,
                                  float* __restrict__ e_f) {
  long long i = (long long)blockIdx.x * blockDim.x + threadIdx.x;
  if (i >= (long long)EN * DN) return;
  int e = (int)(i >> 7), c = (int)(i & 127);
  const float* er = ea + (long long)e * 3;
  e_f[i] = be[c] + er[0] * We[0 * DN + c] + er[1] * We[1 * DN + c] + er[2] * We[2 * DN + c];
}

__global__ void deg_kernel(const int* __restrict__ dst, float* __restrict__ deg) {
  long long e = (long long)blockIdx.x * blockDim.x + threadIdx.x;
  if (e < EN) atomicAdd(&deg[dst[e]], 1.f);
}

__global__ void selfacc_kernel(const float* __restrict__ e_f, const int* __restrict__ dst,
                               float* __restrict__ acc) {
  long long i = (long long)blockIdx.x * blockDim.x + threadIdx.x;
  if (i >= (long long)EN * DN) return;
  int e = (int)(i >> 7), c = (int)(i & 127);
  atomicAdd(&acc[(long long)dst[e] * DN + c], e_f[i]);
}

__global__ void selffin_kernel(const float* __restrict__ acc, const float* __restrict__ deg,
                               float* __restrict__ e_f) {
  long long i = (long long)blockIdx.x * blockDim.x + threadIdx.x;
  if (i >= (long long)NN * DN) return;
  int n = (int)(i >> 7), c = (int)(i & 127);
  e_f[((long long)EN + n) * DN + c] = acc[i] / fmaxf(deg[n], 1.f);
}

__global__ void alpha1_kernel(const float* __restrict__ a_s, const float* __restrict__ a_d,
                              const float* __restrict__ a_e,
                              const int* __restrict__ src, const int* __restrict__ dst,
                              float* __restrict__ alpha, unsigned* __restrict__ m_enc) {
  long long i = (long long)blockIdx.x * blockDim.x + threadIdx.x;
  if (i >= (long long)ENF * HN) return;
  long long e = i >> 3; int hh = (int)(i & 7);
  int s, d;
  if (e < EN) { s = src[e]; d = dst[e]; } else { s = d = (int)(e - EN); }
  float v = a_s[(long long)s * HN + hh] + a_d[(long long)d * HN + hh] + a_e[i];
  v = (v > 0.f) ? v : 0.2f * v;             // leaky relu
  alpha[i] = v;
  atomicMax(&m_enc[(long long)d * HN + hh], fenc(v));
}

__global__ void alpha2_kernel(const int* __restrict__ dst,
                              const unsigned* __restrict__ m_enc,
                              float* __restrict__ alpha, float* __restrict__ ssum) {
  long long i = (long long)blockIdx.x * blockDim.x + threadIdx.x;
  if (i >= (long long)ENF * HN) return;
  long long e = i >> 3; int hh = (int)(i & 7);
  int d = (e < EN) ? dst[e] : (int)(e - EN);
  float v = expf(alpha[i] - fdec(m_enc[(long long)d * HN + hh]));
  alpha[i] = v;
  atomicAdd(&ssum[(long long)d * HN + hh], v);
}

__global__ void scatter_kernel(const int* __restrict__ src, const int* __restrict__ dst,
                               const float* __restrict__ alpha, const float* __restrict__ ssum,
                               const float* __restrict__ hp, float* __restrict__ agg) {
  long long i = (long long)blockIdx.x * blockDim.x + threadIdx.x;
  if (i >= (long long)ENF * HN) return;
  long long e = i >> 3; int hh = (int)(i & 7);
  int s, d;
  if (e < EN) { s = src[e]; d = dst[e]; } else { s = d = (int)(e - EN); }
  float w = alpha[i] / (ssum[(long long)d * HN + hh] + 1e-16f);
  const float* hs = hp + (long long)s * DN + hh * 16;
  float* od = agg + (long long)d * DN + hh * 16;
  for (int c = 0; c < 16; ++c) atomicAdd(&od[c], hs[c] * w);
}

__global__ __launch_bounds__(128) void bnstats_kernel(const float* __restrict__ agg,
                                                      const float* __restrict__ bias,
                                                      float* __restrict__ bnsum,
                                                      float* __restrict__ bnsq) {
  int c = threadIdx.x;
  long long r0 = (long long)blockIdx.x * 256;
  float s = 0.f, q = 0.f;
  for (int r = 0; r < 256; ++r) {
    long long row = r0 + r;
    if (row < NN) { float t = agg[row * DN + c] + bias[c]; s += t; q += t * t; }
  }
  atomicAdd(&bnsum[c], s);
  atomicAdd(&bnsq[c], q);
}

__global__ __launch_bounds__(128) void bnfin_kernel(const float* __restrict__ bnsum,
                                                    const float* __restrict__ bnsq,
                                                    const float* __restrict__ gamma,
                                                    const float* __restrict__ beta,
                                                    float* __restrict__ scale,
                                                    float* __restrict__ shift) {
  int c = threadIdx.x;
  float mu = bnsum[c] / (float)NN;
  float var = bnsq[c] / (float)NN - mu * mu;
  float sc = rsqrtf(var + 1e-5f) * gamma[c];
  scale[c] = sc;
  shift[c] = beta[c] - mu * sc;
}

__global__ void bnapply_kernel(const float* __restrict__ agg, const float* __restrict__ bias,
                               const float* __restrict__ scale, const float* __restrict__ shift,
                               float* __restrict__ res, float* __restrict__ h, int addRes) {
  long long i = (long long)blockIdx.x * blockDim.x + threadIdx.x;
  if (i >= (long long)NN * DN) return;
  int c = (int)(i & 127);
  float y = (agg[i] + bias[c]) * scale[c] + shift[c];
  y = fmaxf(y, 0.f);
  if (addRes) { y += res[i]; res[i] = y; }
  h[i] = y;
}

__global__ void cnt_kernel(const int* __restrict__ batch, float* __restrict__ cnt) {
  long long n = (long long)blockIdx.x * blockDim.x + threadIdx.x;
  if (n < NN) atomicAdd(&cnt[batch[n]], 1.f);
}

__global__ void poolacc_kernel(const float* __restrict__ h, const int* __restrict__ batch,
                               float* __restrict__ gmean, unsigned* __restrict__ gmax) {
  long long i = (long long)blockIdx.x * blockDim.x + threadIdx.x;
  if (i >= (long long)NN * DN) return;
  int n = (int)(i >> 7), c = (int)(i & 127);
  int b = batch[n];
  float v = h[i];
  atomicAdd(&gmean[(long long)b * DN + c], v);
  atomicMax(&gmax[(long long)b * DN + c], fenc(v));
}

// one wave per node: gate = relu(h@W1+b1)@W2 + b2
__global__ __launch_bounds__(256) void gate_kernel(const float* __restrict__ h,
                                                   const float* __restrict__ W1,
                                                   const float* __restrict__ b1,
                                                   const float* __restrict__ W2,
                                                   const float* __restrict__ b2,
                                                   const int* __restrict__ batch,
                                                   float* __restrict__ gate,
                                                   unsigned* __restrict__ gm_enc) {
  int wave = threadIdx.x >> 5;
  int lane = threadIdx.x & 31;
  long long n = (long long)blockIdx.x * 8 + wave;
  if (n >= NN) return;
  const float* hr = h + n * DN;
  float a0 = b1[lane], a1 = b1[lane + 32];
  for (int k = 0; k < DN; ++k) {
    float hv = hr[k];
    a0 += hv * W1[k * 64 + lane];
    a1 += hv * W1[k * 64 + lane + 32];
  }
  float v = fmaxf(a0, 0.f) * W2[lane] + fmaxf(a1, 0.f) * W2[lane + 32];
  for (int o = 16; o >= 1; o >>= 1) v += __shfl_xor(v, o, 32);
  if (lane == 0) {
    float g = v + b2[0];
    gate[n] = g;
    atomicMax(&gm_enc[batch[n]], fenc(g));
  }
}

__global__ void gatenorm_kernel(const float* __restrict__ gate, const int* __restrict__ batch,
                                const unsigned* __restrict__ gm, float* __restrict__ gew,
                                float* __restrict__ gs) {
  long long n = (long long)blockIdx.x * blockDim.x + threadIdx.x;
  if (n >= NN) return;
  int b = batch[n];
  float g = expf(gate[n] - fdec(gm[b]));
  gew[n] = g;
  atomicAdd(&gs[b], g);
}

__global__ void gattacc_kernel(const float* __restrict__ h, const int* __restrict__ batch,
                               const float* __restrict__ gew, const float* __restrict__ gs,
                               float* __restrict__ gatt) {
  long long i = (long long)blockIdx.x * blockDim.x + threadIdx.x;
  if (i >= (long long)NN * DN) return;
  int n = (int)(i >> 7), c = (int)(i & 127);
  int b = batch[n];
  float w = gew[n] / (gs[b] + 1e-16f);
  atomicAdd(&gatt[(long long)b * DN + c], h[i] * w);
}

__global__ void assemble_kernel(const float* __restrict__ gmean, const unsigned* __restrict__ gmax,
                                const float* __restrict__ gatt, const float* __restrict__ cnt,
                                float* __restrict__ gr) {
  long long i = (long long)blockIdx.x * blockDim.x + threadIdx.x;
  if (i >= (long long)GN * 384) return;
  int g = (int)(i / 384), j = (int)(i % 384);
  float v;
  if (j < 128)      v = gmean[(long long)g * DN + j] / fmaxf(cnt[g], 1.f);
  else if (j < 256) v = (cnt[g] > 0.f) ? fdec(gmax[(long long)g * DN + (j - 128)]) : 0.f;
  else              v = gatt[(long long)g * DN + (j - 256)];
  gr[i] = v;
}

__global__ void mlp1_kernel(const float* __restrict__ gr, const float* __restrict__ W1,
                            const float* __restrict__ b1, float* __restrict__ hid) {
  long long i = (long long)blockIdx.x * blockDim.x + threadIdx.x;
  if (i >= (long long)GN * 256) return;
  int g = (int)(i >> 8), j = (int)(i & 255);
  float a = b1[j];
  const float* gg = gr + (long long)g * 384;
  for (int k = 0; k < 384; ++k) a += gg[k] * W1[k * 256 + j];
  hid[i] = fmaxf(a, 0.f);
}

__global__ void mlp2_kernel(const float* __restrict__ hid, const float* __restrict__ W2,
                            const float* __restrict__ b2, float* __restrict__ out) {
  long long i = (long long)blockIdx.x * blockDim.x + threadIdx.x;
  if (i >= (long long)GN * DN) return;
  int g = (int)(i >> 7), j = (int)(i & 127);
  float a = b2[j];
  const float* hh = hid + (long long)g * 256;
  for (int k = 0; k < 256; ++k) a += hh[k] * W2[k * DN + j];
  out[i] = fmaxf(a, 0.f);
}

// ---------------------------------------------------------------------------
extern "C" void kernel_launch(void* const* d_in, const int* in_sizes, int n_in,
                              void* d_out, int out_size, void* d_ws, size_t ws_size,
                              hipStream_t stream) {
  const float* x         = (const float*)d_in[0];
  const float* edge_attr = (const float*)d_in[1];
  const int*   edge_index= (const int*)d_in[2];
  const int*   src       = edge_index;
  const int*   dst       = edge_index + EN;
  const int*   batch     = (const int*)d_in[3];
  const float* W_node = (const float*)d_in[4];  const float* b_node = (const float*)d_in[5];
  const float* W_edge = (const float*)d_in[6];  const float* b_edge = (const float*)d_in[7];
  const float* W_lin  = (const float*)d_in[8];
  const float* att_src= (const float*)d_in[9];  const float* att_dst= (const float*)d_in[10];
  const float* W_eproj= (const float*)d_in[11]; const float* att_edge=(const float*)d_in[12];
  const float* gat_bias=(const float*)d_in[13];
  const float* bn_gamma=(const float*)d_in[14]; const float* bn_beta= (const float*)d_in[15];
  const float* gate_W1= (const float*)d_in[16]; const float* gate_b1= (const float*)d_in[17];
  const float* gate_W2= (const float*)d_in[18]; const float* gate_b2= (const float*)d_in[19];
  const float* mlp_W1 = (const float*)d_in[20]; const float* mlp_b1 = (const float*)d_in[21];
  const float* mlp_W2 = (const float*)d_in[22]; const float* mlp_b2 = (const float*)d_in[23];
  float* out = (float*)d_out;

  // workspace layout (fp32 elements)
  float* p = (float*)d_ws;
  float* h     = p; p += (long long)NN * DN;
  float* res   = p; p += (long long)NN * DN;
  float* e_f   = p; p += (long long)ENF * DN;
  float* hp    = p; p += (long long)NN * DN;
  float* agg   = p; p += (long long)NN * DN;          // also self-loop accumulator
  float* a_s   = p; p += (long long)NN * HN;
  float* a_d   = p; p += (long long)NN * HN;
  float* a_e   = p; p += (long long)ENF * HN;
  float* alpha = p; p += (long long)ENF * HN;
  unsigned* m_enc = (unsigned*)p; p += (long long)NN * HN;
  float* ssum  = p; p += (long long)NN * HN;
  float* deg   = p; p += NN;
  float* bnsum = p; p += DN;
  float* bnsq  = p; p += DN;
  float* bnscale = p; p += DN;
  float* bnshift = p; p += DN;
  float* cnt   = p; p += GN;
  float* gmean = p; p += (long long)GN * DN;
  unsigned* gmax = (unsigned*)p; p += (long long)GN * DN;
  float* gatt  = p; p += (long long)GN * DN;
  float* gatev = p; p += NN;
  float* gew   = p; p += NN;
  unsigned* gm = (unsigned*)p; p += GN;
  float* gs    = p; p += GN;
  float* gr    = p; p += (long long)GN * 384;
  float* hid   = p; p += (long long)GN * 256;

  const int T = 256;
  long long nND  = (long long)NN * DN;
  long long nED  = (long long)EN * DN;
  long long nEH  = (long long)ENF * HN;

  // --- embeddings + self-loop mean edge features ---
  embed_node_kernel<<<CDIV(nND, T), T, 0, stream>>>(x, W_node, b_node, h, res);
  embed_edge_kernel<<<CDIV(nED, T), T, 0, stream>>>(edge_attr, W_edge, b_edge, e_f);
  zero_f32<<<CDIV(nND, T), T, 0, stream>>>(agg, nND);
  zero_f32<<<CDIV((long long)NN, T), T, 0, stream>>>(deg, NN);
  deg_kernel<<<CDIV((long long)EN, T), T, 0, stream>>>(dst, deg);
  selfacc_kernel<<<CDIV(nED, T), T, 0, stream>>>(e_f, dst, agg);
  selffin_kernel<<<CDIV(nND, T), T, 0, stream>>>(agg, deg, e_f);

  // --- GAT layers ---
  for (int l = 0; l < 4; ++l) {
    gemm128_wmma_kernel<<<CDIV((long long)NN, 128), T, 0, stream>>>(
        h, NN, W_lin + (long long)l * DN * DN, hp,
        att_src + l * DN, att_dst + l * DN, a_s, a_d);
    gemm128_wmma_kernel<<<CDIV((long long)ENF, 128), T, 0, stream>>>(
        e_f, ENF, W_eproj + (long long)l * DN * DN, nullptr,
        att_edge + l * DN, nullptr, a_e, nullptr);

    zero_f32<<<CDIV((long long)NN * HN, T), T, 0, stream>>>((float*)m_enc, (long long)NN * HN);
    zero_f32<<<CDIV((long long)NN * HN, T), T, 0, stream>>>(ssum, (long long)NN * HN);
    zero_f32<<<CDIV(nND, T), T, 0, stream>>>(agg, nND);

    alpha1_kernel<<<CDIV(nEH, T), T, 0, stream>>>(a_s, a_d, a_e, src, dst, alpha, m_enc);
    alpha2_kernel<<<CDIV(nEH, T), T, 0, stream>>>(dst, m_enc, alpha, ssum);
    scatter_kernel<<<CDIV(nEH, T), T, 0, stream>>>(src, dst, alpha, ssum, hp, agg);

    zero_f32<<<1, T, 0, stream>>>(bnsum, 2 * DN);   // bnsum + bnsq contiguous
    bnstats_kernel<<<CDIV((long long)NN, 256), 128, 0, stream>>>(agg, gat_bias + l * DN, bnsum, bnsq);
    bnfin_kernel<<<1, 128, 0, stream>>>(bnsum, bnsq, bn_gamma + l * DN, bn_beta + l * DN,
                                        bnscale, bnshift);
    bnapply_kernel<<<CDIV(nND, T), T, 0, stream>>>(agg, gat_bias + l * DN, bnscale, bnshift,
                                                   res, h, (l > 0 && (l % 2 == 0)) ? 1 : 0);
  }

  // --- pooling ---
  zero_f32<<<CDIV((long long)GN, T), T, 0, stream>>>(cnt, GN);
  zero_f32<<<CDIV((long long)GN * DN, T), T, 0, stream>>>(gmean, (long long)GN * DN);
  zero_f32<<<CDIV((long long)GN * DN, T), T, 0, stream>>>((float*)gmax, (long long)GN * DN);
  zero_f32<<<CDIV((long long)GN * DN, T), T, 0, stream>>>(gatt, (long long)GN * DN);
  zero_f32<<<CDIV((long long)GN, T), T, 0, stream>>>((float*)gm, GN);
  zero_f32<<<CDIV((long long)GN, T), T, 0, stream>>>(gs, GN);

  cnt_kernel<<<CDIV((long long)NN, T), T, 0, stream>>>(batch, cnt);
  poolacc_kernel<<<CDIV(nND, T), T, 0, stream>>>(h, batch, gmean, gmax);
  gate_kernel<<<CDIV((long long)NN, 8), T, 0, stream>>>(h, gate_W1, gate_b1, gate_W2, gate_b2,
                                                        batch, gatev, gm);
  gatenorm_kernel<<<CDIV((long long)NN, T), T, 0, stream>>>(gatev, batch, gm, gew, gs);
  gattacc_kernel<<<CDIV(nND, T), T, 0, stream>>>(h, batch, gew, gs, gatt);

  assemble_kernel<<<CDIV((long long)GN * 384, T), T, 0, stream>>>(gmean, gmax, gatt, cnt, gr);
  mlp1_kernel<<<CDIV((long long)GN * 256, T), T, 0, stream>>>(gr, mlp_W1, mlp_b1, hid);
  mlp2_kernel<<<CDIV((long long)GN * DN, T), T, 0, stream>>>(hid, mlp_W2, mlp_b2, out);
}
